// ColWiseGatedAttention_84430467105268
// MI455X (gfx1250) — compile-verified, matrix operand order checked by reading, not run on hardware
//
#include <hip/hip_runtime.h>
#include <math.h>

// ---- problem constants --------------------------------------------------
#define HMDIM 192              // s / t extent
#define WMDIM 192              // i extent
#define CMDIM 256              // c_m
#define NH    8                // heads
#define NC    32               // channels per head
#define NPOS  (HMDIM * WMDIM)  // 36864 spatial positions
#define NTOT  (CMDIM * NPOS)   // 9437184 elements

// ---- bf16 helpers -------------------------------------------------------
typedef __attribute__((ext_vector_type(16))) __bf16 v16bf;
typedef __attribute__((ext_vector_type(8)))  float  v8f;

__device__ __forceinline__ unsigned short f2bf(float f) {
    unsigned u = __float_as_uint(f);
    u += 0x7fffu + ((u >> 16) & 1u);          // round-to-nearest-even
    return (unsigned short)(u >> 16);
}
__device__ __forceinline__ float bf2f(unsigned short h) {
    return __uint_as_float(((unsigned)h) << 16);
}

union FragB {            // one wave-lane's 16 bf16 fragment = two 16B chunks
    v16bf v;
    uint4 q[2];
};

// ---- 1) deterministic global mean/var reduction -------------------------
__global__ void k_reduce1(const float* __restrict__ x, float* __restrict__ part) {
    __shared__ float s1[256], s2[256];
    float a = 0.f, b = 0.f;
    for (int i = blockIdx.x * blockDim.x + threadIdx.x; i < NTOT;
         i += gridDim.x * blockDim.x) {
        float v = x[i];
        a += v; b += v * v;
    }
    s1[threadIdx.x] = a; s2[threadIdx.x] = b;
    __syncthreads();
    for (int off = 128; off > 0; off >>= 1) {
        if (threadIdx.x < (unsigned)off) {
            s1[threadIdx.x] += s1[threadIdx.x + off];
            s2[threadIdx.x] += s2[threadIdx.x + off];
        }
        __syncthreads();
    }
    if (threadIdx.x == 0) {
        part[2 * blockIdx.x + 0] = s1[0];
        part[2 * blockIdx.x + 1] = s2[0];
    }
}

__global__ void k_reduce2(const float* __restrict__ part, float* __restrict__ acc) {
    __shared__ float s1[256], s2[256];
    float a = 0.f, b = 0.f;
    for (int i = threadIdx.x; i < 1024; i += 256) {
        a += part[2 * i + 0];
        b += part[2 * i + 1];
    }
    s1[threadIdx.x] = a; s2[threadIdx.x] = b;
    __syncthreads();
    for (int off = 128; off > 0; off >>= 1) {
        if (threadIdx.x < (unsigned)off) {
            s1[threadIdx.x] += s1[threadIdx.x + off];
            s2[threadIdx.x] += s2[threadIdx.x + off];
        }
        __syncthreads();
    }
    if (threadIdx.x == 0) { acc[0] = s1[0]; acc[1] = s2[0]; }
}

// ---- 2) weights -> bf16 -------------------------------------------------
__global__ void k_cvtw(const float* __restrict__ wq, const float* __restrict__ wk,
                       const float* __restrict__ wv, const float* __restrict__ wg,
                       const float* __restrict__ wr,
                       unsigned short* __restrict__ wall,
                       unsigned short* __restrict__ wrb) {
    int i = blockIdx.x * blockDim.x + threadIdx.x;       // 0..65535
    wall[i]          = f2bf(wq[i]);
    wall[65536 + i]  = f2bf(wk[i]);
    wall[131072 + i] = f2bf(wv[i]);
    wall[196608 + i] = f2bf(wg[i]);
    wrb[i]           = f2bf(wr[i]);
}

// ---- 3) LayerNorm + transpose to x_t[p][c] (bf16) -----------------------
__global__ void k_norm(const float* __restrict__ x, const float* __restrict__ lw,
                       const float* __restrict__ lb, const float* __restrict__ acc,
                       unsigned short* __restrict__ xt) {
    __shared__ float tile[16][17];
    const int p0 = blockIdx.x * 16, c0 = blockIdx.y * 16;
    const float mu   = acc[0] * (1.0f / NTOT);
    const float var  = acc[1] * (1.0f / NTOT) - mu * mu;
    const float rinv = rsqrtf(var + 1e-5f);
    const int tx = threadIdx.x, ty = threadIdx.y;
    const int gi = (c0 + ty) * NPOS + p0 + tx;           // coalesced over p
    tile[tx][ty] = (x[gi] - mu) * rinv * lw[gi] + lb[gi];
    __syncthreads();
    // x_t[p][c], p = p0+ty, c = c0+tx  (coalesced over c)
    xt[(size_t)(p0 + ty) * CMDIM + c0 + tx] = f2bf(tile[ty][tx]);
}

// ---- 4) four projection GEMMs: Y = W(256x256) * X(256x36864) ------------
// grid = (2304, 2, 4), block = (32, 8): one 16x16 tile per wave
__global__ void k_proj(const unsigned short* __restrict__ wall,
                       const unsigned short* __restrict__ xt,
                       const float* __restrict__ bg,
                       unsigned short* __restrict__ qw,
                       unsigned short* __restrict__ kw,
                       unsigned short* __restrict__ vw,
                       unsigned short* __restrict__ gw) {
    const int lane = threadIdx.x, wave = threadIdx.y;
    const int l16 = lane & 15, half = lane >> 4;
    const int mtile = blockIdx.y * 8 + wave;
    const int ntile = blockIdx.x;
    const int proj  = blockIdx.z;

    const unsigned short* wrow = wall + proj * 65536 + (mtile * 16 + l16) * CMDIM;
    const unsigned short* xrow = xt + (size_t)(ntile * 16 + l16) * CMDIM;

    v8f acc = {};
#pragma unroll
    for (int ks = 0; ks < 8; ++ks) {
        const int kb = ks * 32 + half * 8;
        FragB a, b;
        a.q[0] = *(const uint4*)(wrow + kb);
        a.q[1] = *(const uint4*)(wrow + kb + 16);
        b.q[0] = *(const uint4*)(xrow + kb);
        b.q[1] = *(const uint4*)(xrow + kb + 16);
        acc = __builtin_amdgcn_wmma_f32_16x16x32_bf16(false, a.v, false, b.v,
                                                      (short)0, acc, false, false);
    }

    const int p = ntile * 16 + l16;
    const int s = p / WMDIM, col = p % WMDIM;
#pragma unroll
    for (int r = 0; r < 8; ++r) {
        const int o = mtile * 16 + r + 8 * half;         // output channel h*32+c
        const int h = o >> 5, c = o & 31;
        float val = acc[r];
        if (proj == 0) {                                 // q[h][i][s][c]
            qw[((size_t)(h * WMDIM + col) * HMDIM + s) * NC + c] = f2bf(val);
        } else if (proj == 1) {                          // k[h][i][t][c]
            kw[((size_t)(h * WMDIM + col) * HMDIM + s) * NC + c] = f2bf(val);
        } else if (proj == 2) {                          // v[h][s][c][t]
            vw[((size_t)(h * HMDIM + s) * NC + c) * WMDIM + col] = f2bf(val);
        } else {                                         // g[h][s][c][i] = sigmoid
            val = 1.0f / (1.0f + __expf(-(val + bg[o])));
            gw[((size_t)(h * HMDIM + s) * NC + c) * WMDIM + col] = f2bf(val);
        }
    }
}

// ---- 5) logits + softmax, per (h, i): a[s,t] ----------------------------
// grid = (192, 8), block = (32, 12): wave w owns s-tile w (16 rows), 12 t-tiles
__global__ void k_logits(const unsigned short* __restrict__ qw,
                         const unsigned short* __restrict__ kw,
                         unsigned short* __restrict__ aw) {
    const int lane = threadIdx.x, wave = threadIdx.y;
    const int l16 = lane & 15, half = lane >> 4;
    const int i = blockIdx.x, h = blockIdx.y;
    const int s0 = wave * 16;

    const unsigned short* qbase = qw + (size_t)(h * WMDIM + i) * HMDIM * NC;
    const unsigned short* kbase = kw + (size_t)(h * WMDIM + i) * HMDIM * NC;

    FragB qa;
    const unsigned short* qrow = qbase + (s0 + l16) * NC;
    qa.q[0] = *(const uint4*)(qrow + half * 8);
    qa.q[1] = *(const uint4*)(qrow + 16 + half * 8);

    v8f acc[12];
#pragma unroll
    for (int tt = 0; tt < 12; ++tt) {
        FragB kb;
        const unsigned short* krow = kbase + (tt * 16 + l16) * NC;
        kb.q[0] = *(const uint4*)(krow + half * 8);
        kb.q[1] = *(const uint4*)(krow + 16 + half * 8);
        v8f z = {};
        acc[tt] = __builtin_amdgcn_wmma_f32_16x16x32_bf16(false, qa.v, false, kb.v,
                                                          (short)0, z, false, false);
    }

    const float scale = 0.17677669529663687f;            // 1/sqrt(32)
#pragma unroll
    for (int r = 0; r < 8; ++r) {
        float mx = -3.4e38f;
#pragma unroll
        for (int tt = 0; tt < 12; ++tt) mx = fmaxf(mx, acc[tt][r]);
        for (int m = 1; m < 16; m <<= 1) mx = fmaxf(mx, __shfl_xor(mx, m, 32));
        float sum = 0.f;
#pragma unroll
        for (int tt = 0; tt < 12; ++tt) {
            float e = __expf((acc[tt][r] - mx) * scale);
            acc[tt][r] = e; sum += e;
        }
        for (int m = 1; m < 16; m <<= 1) sum += __shfl_xor(sum, m, 32);
        const float inv = 1.0f / sum;
        const int s = s0 + r + 8 * half;
        unsigned short* arow = aw + ((size_t)(h * HMDIM + s) * WMDIM + i) * HMDIM;
#pragma unroll
        for (int tt = 0; tt < 12; ++tt)
            arow[tt * 16 + l16] = f2bf(acc[tt][r] * inv);
    }
}

// ---- 6) O = V(32x192) * A(192x192), gate, pack o_t[p][c*8+h] ------------
// grid = (192, 8) = (s, h), block = (32, 12): wave = i-tile
__global__ void k_attno(const unsigned short* __restrict__ vw,
                        const unsigned short* __restrict__ aw,
                        const unsigned short* __restrict__ gw,
                        unsigned short* __restrict__ ot) {
    const int lane = threadIdx.x, wave = threadIdx.y;
    const int l16 = lane & 15, half = lane >> 4;
    const int s = blockIdx.x, h = blockIdx.y;

    const unsigned short* vbase = vw + (size_t)(h * HMDIM + s) * NC * WMDIM;
    const unsigned short* abase = aw + (size_t)(h * HMDIM + s) * WMDIM * HMDIM;
    const unsigned short* arow  = abase + (wave * 16 + l16) * HMDIM;
    const unsigned short* vrow0 = vbase + l16 * WMDIM;
    const unsigned short* vrow1 = vbase + (16 + l16) * WMDIM;

    v8f acc0 = {}, acc1 = {};
#pragma unroll
    for (int ks = 0; ks < 6; ++ks) {
        const int kb = ks * 32 + half * 8;
        FragB b, a0, a1;
        b.q[0]  = *(const uint4*)(arow + kb);
        b.q[1]  = *(const uint4*)(arow + kb + 16);
        a0.q[0] = *(const uint4*)(vrow0 + kb);
        a0.q[1] = *(const uint4*)(vrow0 + kb + 16);
        a1.q[0] = *(const uint4*)(vrow1 + kb);
        a1.q[1] = *(const uint4*)(vrow1 + kb + 16);
        acc0 = __builtin_amdgcn_wmma_f32_16x16x32_bf16(false, a0.v, false, b.v,
                                                       (short)0, acc0, false, false);
        acc1 = __builtin_amdgcn_wmma_f32_16x16x32_bf16(false, a1.v, false, b.v,
                                                       (short)0, acc1, false, false);
    }

    const int i = wave * 16 + l16;
    const unsigned short* gbase = gw + (size_t)(h * HMDIM + s) * NC * WMDIM;
#pragma unroll
    for (int r = 0; r < 8; ++r) {
        const int c0 = r + 8 * half;
        const float g0 = bf2f(gbase[c0 * WMDIM + i]);
        const float g1 = bf2f(gbase[(16 + c0) * WMDIM + i]);
        // reference output packing: o_flat[c*H + h]
        ot[(size_t)(s * WMDIM + i) * CMDIM + c0 * NH + h]        = f2bf(acc0[r] * g0);
        ot[(size_t)(s * WMDIM + i) * CMDIM + (16 + c0) * NH + h] = f2bf(acc1[r] * g1);
    }
}

// ---- 7) output projection: out = Wr(256x256) * o_flat + br (fp32 out) ---
// grid = (2304, 2), block = (32, 8)
__global__ void k_final(const unsigned short* __restrict__ wrb,
                        const unsigned short* __restrict__ ot,
                        const float* __restrict__ br,
                        float* __restrict__ out) {
    const int lane = threadIdx.x, wave = threadIdx.y;
    const int l16 = lane & 15, half = lane >> 4;
    const int mtile = blockIdx.y * 8 + wave;
    const int ntile = blockIdx.x;

    const unsigned short* wrow = wrb + (mtile * 16 + l16) * CMDIM;
    const unsigned short* xrow = ot + (size_t)(ntile * 16 + l16) * CMDIM;

    v8f acc = {};
#pragma unroll
    for (int ks = 0; ks < 8; ++ks) {
        const int kb = ks * 32 + half * 8;
        FragB a, b;
        a.q[0] = *(const uint4*)(wrow + kb);
        a.q[1] = *(const uint4*)(wrow + kb + 16);
        b.q[0] = *(const uint4*)(xrow + kb);
        b.q[1] = *(const uint4*)(xrow + kb + 16);
        acc = __builtin_amdgcn_wmma_f32_16x16x32_bf16(false, a.v, false, b.v,
                                                      (short)0, acc, false, false);
    }

    const int p = ntile * 16 + l16;
#pragma unroll
    for (int r = 0; r < 8; ++r) {
        const int m = mtile * 16 + r + 8 * half;
        out[(size_t)m * NPOS + p] = acc[r] + br[m];
    }
}

// ---- host launcher ------------------------------------------------------
extern "C" void kernel_launch(void* const* d_in, const int* in_sizes, int n_in,
                              void* d_out, int out_size, void* d_ws, size_t ws_size,
                              hipStream_t stream) {
    (void)in_sizes; (void)n_in; (void)out_size; (void)ws_size;
    const float* msa = (const float*)d_in[0];
    const float* lnw = (const float*)d_in[1];
    const float* lnb = (const float*)d_in[2];
    const float* wq  = (const float*)d_in[3];
    const float* wk  = (const float*)d_in[4];
    const float* wv  = (const float*)d_in[5];
    const float* wg  = (const float*)d_in[6];
    const float* bg  = (const float*)d_in[7];
    const float* wr  = (const float*)d_in[8];
    const float* br  = (const float*)d_in[9];
    float* out = (float*)d_out;

    char* p = (char*)d_ws;
    auto alloc = [&](size_t bytes) -> char* {
        char* r = p;
        p += (bytes + 255) & ~(size_t)255;
        return r;
    };
    float*          acc  = (float*)         alloc(2 * sizeof(float));
    float*          part = (float*)         alloc(2 * 1024 * sizeof(float));
    unsigned short* wall = (unsigned short*)alloc(4 * 65536 * 2);
    unsigned short* wrb  = (unsigned short*)alloc(65536 * 2);
    unsigned short* xt   = (unsigned short*)alloc((size_t)NPOS * CMDIM * 2);
    unsigned short* qw   = (unsigned short*)alloc((size_t)NH * WMDIM * HMDIM * NC * 2);
    unsigned short* kw   = (unsigned short*)alloc((size_t)NH * WMDIM * HMDIM * NC * 2);
    unsigned short* vw   = (unsigned short*)alloc((size_t)NH * HMDIM * NC * WMDIM * 2);
    unsigned short* gw   = (unsigned short*)alloc((size_t)NH * HMDIM * NC * WMDIM * 2);
    unsigned short* aw   = (unsigned short*)alloc((size_t)NH * HMDIM * WMDIM * HMDIM * 2);
    unsigned short* ot   = (unsigned short*)alloc((size_t)NPOS * CMDIM * 2);

    k_reduce1<<<1024, 256, 0, stream>>>(msa, part);
    k_reduce2<<<1, 256, 0, stream>>>(part, acc);
    k_cvtw<<<256, 256, 0, stream>>>(wq, wk, wv, wg, wr, wall, wrb);
    k_norm<<<dim3(NPOS / 16, CMDIM / 16), dim3(16, 16), 0, stream>>>(msa, lnw, lnb, acc, xt);
    k_proj<<<dim3(NPOS / 16, 2, 4), dim3(32, 8), 0, stream>>>(wall, xt, bg, qw, kw, vw, gw);
    k_logits<<<dim3(WMDIM, NH), dim3(32, 12), 0, stream>>>(qw, kw, aw);
    k_attno<<<dim3(HMDIM, NH), dim3(32, 12), 0, stream>>>(vw, aw, gw, ot);
    k_final<<<dim3(NPOS / 16, 2), dim3(32, 8), 0, stream>>>(wrb, ot, br, out);
}
